// Social_Model_4518305595964
// MI455X (gfx1250) — matrix operand order repaired and checked by reading, not compiled, not run on hardware
//
#include <hip/hip_runtime.h>
#include <hip/hip_bf16.h>
#include <math.h>

typedef __attribute__((ext_vector_type(16))) _Float16 v16h;
typedef __attribute__((ext_vector_type(8)))  float    v8f;

#define B_SZ   2048
#define NMAX   16
#define T_SZ   50
#define H_ENC  64
#define G_ENC  256   // 4*H_ENC
#define H_DEC  128
#define G_DEC  512   // 4*H_DEC
#define PRED   30

// Branch-free activations on the gfx1250 transcendental unit.
// V_TANH_F32 is a CDNA5/gfx1250 trans op (1 instr); sigmoid rides it.
#if __has_builtin(__builtin_amdgcn_tanhf)
__device__ __forceinline__ float fast_tanh(float x) {
    return __builtin_amdgcn_tanhf(x);
}
#else
__device__ __forceinline__ float fast_tanh(float x) {
    // branch-free fallback: tanh(x) = 1 - 2/(exp(2x)+1)
    return 1.0f - 2.0f / (__expf(2.0f * x) + 1.0f);
}
#endif

__device__ __forceinline__ float sigm(float x) {
    return fmaf(0.5f, fast_tanh(0.5f * x), 0.5f);
}

// ---------------------------------------------------------------------------
// Encoder: batched LSTM(2 -> 64), returns last hidden state.
// One wave handles 16 sequences; block = 4 waves = 64 sequences.
// Whh is staged into LDS as f16 in WMMA B-fragment order.
// Per step: gates[16x256] = xproj + h[16x64] @ Whh^T via v_wmma_f32_16x16x32_f16.
// ---------------------------------------------------------------------------
__global__ __launch_bounds__(128) void lstm_enc_kernel(
    const float* __restrict__ x,      // [N, T, 2]
    const float* __restrict__ Wih,    // [256, 2]
    const float* __restrict__ Whh,    // [256, 64]
    const float* __restrict__ bih,    // [256]
    const float* __restrict__ bhh,    // [256]
    float* __restrict__ hout,         // [N, 64]
    int T)
{
    // B fragments: 16 n-tiles * 2 k-chunks * 32 lanes * 16 halves = 32 KB
    __shared__ __align__(32) _Float16 sB[16 * 2 * 32 * 16];
    __shared__ float sBias[G_ENC];
    __shared__ float sWih[G_ENC * 2];
    __shared__ float sX[4][16][2];
    __shared__ __align__(8) float sH[4][16 * H_ENC];   // per-wave h tile, 16 KB

    const int tid  = threadIdx.x;
    const int wave = tid >> 5;
    const int lane = tid & 31;
    const int grp  = lane >> 4;           // 0: lanes 0-15, 1: lanes 16-31
    const int lm   = lane & 15;
    const int seqBase = blockIdx.x * 64 + wave * 16;

    // Stage Whh as f16 in B-fragment order:
    //  element (k, n) of Whh^T == Whh[n*64 + k]
    //  frag = tile*2 + kc; lane ln; vgpr v; half e
    //  n = tile*16 + (ln&15);  k = kc*32 + (ln>>4)*16 + v*2 + e
    for (int i = tid; i < 16 * 2 * 32 * 16; i += 128) {
        int e  = i & 1;
        int v  = (i >> 1) & 7;
        int ln = (i >> 4) & 31;
        int kc = (i >> 9) & 1;
        int tl = i >> 10;
        int n  = tl * 16 + (ln & 15);
        int k  = kc * 32 + (ln >> 4) * 16 + v * 2 + e;
        sB[i] = (_Float16)Whh[n * H_ENC + k];
    }
    for (int i = tid; i < G_ENC; i += 128) {
        sBias[i]        = bih[i] + bhh[i];
        sWih[i * 2]     = Wih[i * 2];
        sWih[i * 2 + 1] = Wih[i * 2 + 1];
    }
    __syncthreads();

    // c fragments (persistent, WMMA C/D layout): 4 j-tiles of 16
    v8f cfr[4];
    for (int t = 0; t < 4; ++t) cfr[t] = (v8f){};
    // h0 = 0
    for (int i = lane; i < 16 * H_ENC; i += 32) sH[wave][i] = 0.0f;

    for (int step = 0; step < T; ++step) {
        // stage x[:, step, :] for this wave's 16 sequences (wave-private LDS,
        // per-wave LDS ops are in-order -> no block barrier needed)
        if (lane < 16) {
            int s = seqBase + lane;
            sX[wave][lane][0] = x[(s * T + step) * 2 + 0];
            sX[wave][lane][1] = x[(s * T + step) * 2 + 1];
        }

        // Build A fragments (16x32 f16) from h, matching B-fragment k order:
        //  lane grp 0: v0..7 -> K 0..15
        //  lane grp 1: v0..7 -> K 16..31   (+ kc*32)
        v16h afr[2];
        for (int kc = 0; kc < 2; ++kc) {
            for (int v = 0; v < 8; ++v) {
                int k = kc * 32 + grp * 16 + v * 2;
                float2 hv = *(const float2*)&sH[wave][lm * H_ENC + k];
                afr[kc][v * 2 + 0] = (_Float16)hv.x;
                afr[kc][v * 2 + 1] = (_Float16)hv.y;
            }
        }

        // Gate tiles: j-tile jt covers hidden cols [jt*16, jt*16+16);
        // gate g lives at n-tile g*4 + jt  (torch order i,f,g,o)
        for (int jt = 0; jt < 4; ++jt) {
            v8f acc[4];
            // init accumulators with x-projection + bias (K=2, VALU)
            for (int gs = 0; gs < 4; ++gs) {
                int n = (gs * 4 + jt) * 16 + lm;
                float w0 = sWih[n * 2], w1 = sWih[n * 2 + 1], bs = sBias[n];
                v8f a;
                for (int r = 0; r < 8; ++r) {
                    int m = r + grp * 8;
                    a[r] = fmaf(sX[wave][m][0], w0,
                           fmaf(sX[wave][m][1], w1, bs));
                }
                acc[gs] = a;
            }
            // WMMA: gates += h @ Whh^T
            for (int gs = 0; gs < 4; ++gs) {
                int tl = gs * 4 + jt;
                for (int kc = 0; kc < 2; ++kc) {
                    const v16h bfr =
                        *(const v16h*)&sB[((tl * 2 + kc) * 32 + lane) * 16];
                    acc[gs] = __builtin_amdgcn_wmma_f32_16x16x32_f16(
                        false, afr[kc], false, bfr, (short)0, acc[gs],
                        false, false);
                }
            }
            // elementwise LSTM update in fragment layout (branch-free)
            v8f c = cfr[jt];
            v8f hnew;
            for (int r = 0; r < 8; ++r) {
                float ig = sigm(acc[0][r]);
                float fg = sigm(acc[1][r]);
                float gg = fast_tanh(acc[2][r]);
                float og = sigm(acc[3][r]);
                float cn = fmaf(fg, c[r], ig * gg);
                c[r] = cn;
                hnew[r] = og * fast_tanh(cn);
            }
            cfr[jt] = c;
            for (int r = 0; r < 8; ++r) {
                int m = r + grp * 8;
                sH[wave][m * H_ENC + jt * 16 + lm] = hnew[r];
            }
        }
    }

    // write last hidden state, coalesced from LDS
    for (int i = lane; i < 16 * H_ENC; i += 32) {
        int m = i / H_ENC, j = i % H_ENC;
        hout[(seqBase + m) * H_ENC + j] = sH[wave][i];
    }
}

// ---------------------------------------------------------------------------
// Neighbour max-pool over valid entries + concat with agent embedding -> h0
// ---------------------------------------------------------------------------
__global__ __launch_bounds__(256) void pool_concat_kernel(
    const float* __restrict__ agent,   // [B, 64]
    const float* __restrict__ nenc,    // [B, 16, 64]
    const int*   __restrict__ cnt,     // [B]
    float* __restrict__ h0)            // [B, 128]
{
    int i = blockIdx.x * 256 + threadIdx.x;   // over B*64
    if (i >= B_SZ * H_ENC) return;
    int b = i >> 6, j = i & 63;
    int c = cnt[b];
    if (c > NMAX) c = NMAX;
    float m = 0.0f;
    if (c > 0) {
        m = -INFINITY;
        for (int nn = 0; nn < c; ++nn)
            m = fmaxf(m, nenc[(b * NMAX + nn) * H_ENC + j]);
    }
    h0[b * H_DEC + j]         = agent[i];
    h0[b * H_DEC + H_ENC + j] = m;
}

// ---------------------------------------------------------------------------
// Decoder weight prep: W = Wih_d + Whh_d (input==hidden), f16, B-fragment
// order: 32 n-tiles * 4 k-chunks; n = tl*16 + (ln&15), k = kc*32+(ln>>4)*16+2v+e
// ---------------------------------------------------------------------------
__global__ __launch_bounds__(256) void prep_dec_weights_kernel(
    const float* __restrict__ Wih_d,   // [512, 128]
    const float* __restrict__ Whh_d,   // [512, 128]
    _Float16* __restrict__ wB)         // [512*128] halves in frag order
{
    int i = blockIdx.x * 256 + threadIdx.x;
    if (i >= G_DEC * H_DEC) return;
    int e  = i & 1;
    int v  = (i >> 1) & 7;
    int ln = (i >> 4) & 31;
    int kc = (i >> 9) & 3;
    int tl = i >> 11;
    int n  = tl * 16 + (ln & 15);
    int k  = kc * 32 + (ln >> 4) * 16 + v * 2 + e;
    wB[i] = (_Float16)(Wih_d[n * H_DEC + k] + Whh_d[n * H_DEC + k]);
}

// ---------------------------------------------------------------------------
// Decoder: 30 steps of LSTMCell(128->128) with input==hidden, + pos head.
// One wave handles 16 rows; weights streamed from L2-resident scratch.
// ---------------------------------------------------------------------------
__global__ __launch_bounds__(128) void lstm_dec_kernel(
    const float* __restrict__ h0,      // [B, 128]
    const _Float16* __restrict__ wB,   // frag-ordered f16 combined weights
    const float* __restrict__ bihd, const float* __restrict__ bhhd, // [512]
    const float* __restrict__ Wpos,    // [2, 128]
    const float* __restrict__ bpos,    // [2]
    float* __restrict__ out)           // [B, 30, 2]
{
    __shared__ __align__(16) float sH[4][16 * H_DEC];   // 32 KB
    __shared__ float sBias[G_DEC];
    __shared__ __align__(16) float sWpos[2 * H_DEC];
    __shared__ float sBpos[2];

    const int tid  = threadIdx.x;
    const int wave = tid >> 5;
    const int lane = tid & 31;
    const int grp  = lane >> 4;
    const int lm   = lane & 15;
    const int rowBase = blockIdx.x * 64 + wave * 16;

    for (int i = tid; i < G_DEC; i += 128) sBias[i] = bihd[i] + bhhd[i];
    for (int i = tid; i < 2 * H_DEC; i += 128) sWpos[i] = Wpos[i];
    if (tid < 2) sBpos[tid] = bpos[tid];
    // load h0 (wave-private region)
    for (int i = lane; i < 16 * H_DEC; i += 32)
        sH[wave][i] = h0[(rowBase + i / H_DEC) * H_DEC + i % H_DEC];
    __syncthreads();

    v8f cfr[8];
    for (int t = 0; t < 8; ++t) cfr[t] = (v8f){};

    for (int step = 0; step < PRED; ++step) {
        // A fragments: K=128 in 4 chunks of 32 (k = kc*32 + grp*16 + 2v + e, matching wB)
        v16h afr[4];
        for (int kc = 0; kc < 4; ++kc) {
            for (int v = 0; v < 8; ++v) {
                int k = kc * 32 + grp * 16 + v * 2;
                float2 hv = *(const float2*)&sH[wave][lm * H_DEC + k];
                afr[kc][v * 2 + 0] = (_Float16)hv.x;
                afr[kc][v * 2 + 1] = (_Float16)hv.y;
            }
        }

        for (int jt = 0; jt < 8; ++jt) {
            v8f acc[4];
            for (int gs = 0; gs < 4; ++gs) {
                int n = gs * H_DEC + jt * 16 + lm;
                float bs = sBias[n];
                v8f a;
                for (int r = 0; r < 8; ++r) a[r] = bs;
                acc[gs] = a;
            }
            for (int gs = 0; gs < 4; ++gs) {
                int tl = gs * 8 + jt;
                for (int kc = 0; kc < 4; ++kc) {
                    int f = tl * 4 + kc;
                    const v16h bfr = *(const v16h*)&wB[(f * 32 + lane) * 16];
                    acc[gs] = __builtin_amdgcn_wmma_f32_16x16x32_f16(
                        false, afr[kc], false, bfr, (short)0, acc[gs],
                        false, false);
                }
            }
            v8f c = cfr[jt];
            v8f hnew;
            for (int r = 0; r < 8; ++r) {
                float ig = sigm(acc[0][r]);
                float fg = sigm(acc[1][r]);
                float gg = fast_tanh(acc[2][r]);
                float og = sigm(acc[3][r]);
                float cn = fmaf(fg, c[r], ig * gg);
                c[r] = cn;
                hnew[r] = og * fast_tanh(cn);
            }
            cfr[jt] = c;
            for (int r = 0; r < 8; ++r) {
                int m = r + grp * 8;
                sH[wave][m * H_DEC + jt * 16 + lm] = hnew[r];
            }
        }

        // position head: lane (grp,lm) computes out component grp of row lm
        {
            const float4* hp = (const float4*)&sH[wave][lm * H_DEC];
            const float4* wp = (const float4*)&sWpos[grp * H_DEC];
            float sum = sBpos[grp];
            for (int k = 0; k < H_DEC / 4; ++k) {
                float4 hv = hp[k], wv = wp[k];
                sum += hv.x * wv.x + hv.y * wv.y + hv.z * wv.z + hv.w * wv.w;
            }
            out[((rowBase + lm) * PRED + step) * 2 + grp] = sum;
        }
    }
}

// ---------------------------------------------------------------------------
extern "C" void kernel_launch(void* const* d_in, const int* in_sizes, int n_in,
                              void* d_out, int out_size, void* d_ws, size_t ws_size,
                              hipStream_t stream)
{
    const float* train_agent = (const float*)d_in[0];
    const float* neighbour   = (const float*)d_in[1];
    const int*   ncnt        = (const int*)  d_in[2];
    const float* Wih_a = (const float*)d_in[3];
    const float* Whh_a = (const float*)d_in[4];
    const float* bih_a = (const float*)d_in[5];
    const float* bhh_a = (const float*)d_in[6];
    const float* Wih_n = (const float*)d_in[7];
    const float* Whh_n = (const float*)d_in[8];
    const float* bih_n = (const float*)d_in[9];
    const float* bhh_n = (const float*)d_in[10];
    const float* Wih_d = (const float*)d_in[11];
    const float* Whh_d = (const float*)d_in[12];
    const float* bih_d = (const float*)d_in[13];
    const float* bhh_d = (const float*)d_in[14];
    const float* Wpos  = (const float*)d_in[15];
    const float* bpos  = (const float*)d_in[16];
    float* out = (float*)d_out;

    // Workspace layout (floats), 32B-aligned segments:
    float* agent_emb = (float*)d_ws;                            // 2048*64
    float* n_enc     = agent_emb + B_SZ * H_ENC;                // 32768*64
    float* h0        = n_enc + B_SZ * NMAX * H_ENC;             // 2048*128
    _Float16* wB     = (_Float16*)(h0 + B_SZ * H_DEC);          // 512*128 halves

    // 1) agent encoder: N = 2048 -> 32 blocks of 64 sequences
    lstm_enc_kernel<<<B_SZ / 64, 128, 0, stream>>>(
        train_agent, Wih_a, Whh_a, bih_a, bhh_a, agent_emb, T_SZ);

    // 2) neighbour encoder: N = 2048*16 = 32768 -> 512 blocks
    lstm_enc_kernel<<<(B_SZ * NMAX) / 64, 128, 0, stream>>>(
        neighbour, Wih_n, Whh_n, bih_n, bhh_n, n_enc, T_SZ);

    // 3) decoder weight prep (independent of 1/2 -> can overlap)
    prep_dec_weights_kernel<<<(G_DEC * H_DEC + 255) / 256, 256, 0, stream>>>(
        Wih_d, Whh_d, wB);

    // 4) pool + concat -> h0
    pool_concat_kernel<<<(B_SZ * H_ENC + 255) / 256, 256, 0, stream>>>(
        agent_emb, n_enc, ncnt, h0);

    // 5) decoder rollout
    lstm_dec_kernel<<<B_SZ / 64, 128, 0, stream>>>(
        h0, wB, bih_d, bhh_d, Wpos, bpos, out);
}
